// NGramAttention_45002667327809
// MI455X (gfx1250) — compile-verified, matrix operand order checked by reading, not run on hardware
//
#include <hip/hip_runtime.h>
#include <math.h>

typedef __attribute__((ext_vector_type(16))) _Float16 v16h;
typedef __attribute__((ext_vector_type(8)))  float    v8f;

static constexpr int Bn = 32;
static constexpr int Qn = 512;
static constexpr int Mn = 512;
static constexpr int Dn = 1024;
static constexpr int ORDERS = 3;

// ============================ helpers ============================
union H4 { unsigned long long u; _Float16 h[4]; };

// ---------------- f32 -> f16 conversion (vectorized) ----------------
__global__ void k_cvt_f16(const float* __restrict__ x, _Float16* __restrict__ y, long long n4)
{
    long long i = (long long)blockIdx.x * blockDim.x + threadIdx.x;
    if (i >= n4) return;
    float4 v = reinterpret_cast<const float4*>(x)[i];
    H4 p;
    p.h[0] = (_Float16)v.x; p.h[1] = (_Float16)v.y;
    p.h[2] = (_Float16)v.z; p.h[3] = (_Float16)v.w;
    reinterpret_cast<unsigned long long*>(y)[i] = p.u;
}

// ---------------- tanh(f32) -> f16 ----------------
__global__ void k_tanh_f16(const float* __restrict__ x, _Float16* __restrict__ y, long long n4)
{
    long long i = (long long)blockIdx.x * blockDim.x + threadIdx.x;
    if (i >= n4) return;
    float4 v = reinterpret_cast<const float4*>(x)[i];
    H4 p;
    p.h[0] = (_Float16)tanhf(v.x); p.h[1] = (_Float16)tanhf(v.y);
    p.h[2] = (_Float16)tanhf(v.z); p.h[3] = (_Float16)tanhf(v.w);
    reinterpret_cast<unsigned long long*>(y)[i] = p.u;
}

// ---------------- batched f16 [R,C] -> [C,R] transpose (LDS tiled) ----------------
__global__ __launch_bounds__(256)
void k_tr_f16(const _Float16* __restrict__ src, _Float16* __restrict__ dst,
              int R, int C, long long sStride, long long dStride)
{
    __shared__ _Float16 t[32][33];
    src += (long long)blockIdx.z * sStride;
    dst += (long long)blockIdx.z * dStride;
    int tx = threadIdx.x, ty = threadIdx.y;            // (32, 8)
#pragma unroll
    for (int k = 0; k < 4; ++k) {
        int r = blockIdx.y * 32 + ty + k * 8;
        int c = blockIdx.x * 32 + tx;
        t[ty + k * 8][tx] = src[(long long)r * C + c];
    }
    __syncthreads();
#pragma unroll
    for (int k = 0; k < 4; ++k) {
        int c = blockIdx.x * 32 + ty + k * 8;          // dst row = src col
        int r = blockIdx.y * 32 + tx;                  // dst col = src row
        dst[(long long)c * R + r] = t[tx][ty + k * 8];
    }
}

static void run_tr(const _Float16* s, _Float16* d, int R, int C, int batches, hipStream_t st)
{
    dim3 g((unsigned)(C / 32), (unsigned)(R / 32), (unsigned)batches), b(32, 8);
    k_tr_f16<<<g, b, 0, st>>>(s, d, R, C, (long long)R * C, (long long)R * C);
}

// ---------------- n-gram embedding build: [B,512,2D] f16, zero-padded ----------------
__global__ void k_ngram_build(const float* __restrict__ mem, _Float16* __restrict__ ng,
                              int order, int nvalid)
{
    long long t = (long long)blockIdx.x * blockDim.x + threadIdx.x;
    int d4 = (int)(t & 255);          // Dn/4 = 256
    int n  = (int)((t >> 8) & 511);
    int b  = (int)(t >> 17);
    if (b >= Bn) return;
    int d = d4 * 4;
    H4 pm, px;
    if (n < nvalid) {
        const float* base = mem + ((long long)b * Mn + n) * Dn + d;
        float4 s  = *reinterpret_cast<const float4*>(base);
        float4 mx = s;
        for (int j = 1; j < order; ++j) {
            float4 v = *reinterpret_cast<const float4*>(base + (long long)j * Dn);
            s.x += v.x; s.y += v.y; s.z += v.z; s.w += v.w;
            mx.x = fmaxf(mx.x, v.x); mx.y = fmaxf(mx.y, v.y);
            mx.z = fmaxf(mx.z, v.z); mx.w = fmaxf(mx.w, v.w);
        }
        float inv = 1.0f / (float)order;
        pm.h[0] = (_Float16)(s.x * inv); pm.h[1] = (_Float16)(s.y * inv);
        pm.h[2] = (_Float16)(s.z * inv); pm.h[3] = (_Float16)(s.w * inv);
        px.h[0] = (_Float16)mx.x; px.h[1] = (_Float16)mx.y;
        px.h[2] = (_Float16)mx.z; px.h[3] = (_Float16)mx.w;
    } else {
        pm.u = 0ULL; px.u = 0ULL;
    }
    long long o = ((long long)b * 512 + n) * 2048;
    *reinterpret_cast<unsigned long long*>(ng + o + d)        = pm.u;
    *reinterpret_cast<unsigned long long*>(ng + o + 1024 + d) = px.u;
}

// ---------------- masked softmax: one wave32 per row of 512 logits ----------------
__global__ __launch_bounds__(256)
void k_softmax(const float* __restrict__ logits, const int* __restrict__ mlens,
               float* __restrict__ out_align, _Float16* __restrict__ av_h,
               int order, int nvalid)
{
    int lane = threadIdx.x & 31;
    int wrow = threadIdx.x >> 5;
    long long r = (long long)blockIdx.x * 8 + wrow;   // row in [0, B*Q)
    int b  = (int)(r >> 9);
    int qq = (int)(r & 511);
    int mlen = mlens[b];
    int len  = (order == 0) ? mlen : (mlen - order + 1);
    float ind = (order == 0) ? 1.0f : ((mlen >= order) ? 1.0f : 0.0f);

    const float* row = logits + r * 512;
    float vals[16];
    float mx = -INFINITY;
#pragma unroll
    for (int i = 0; i < 16; ++i) {
        int j = lane + i * 32;
        bool ok = (j < nvalid) && ((j < len) || (j == 0 && len <= 0));
        float x = ok ? row[j] : -INFINITY;
        vals[i] = x;
        mx = fmaxf(mx, x);
    }
    for (int o = 16; o > 0; o >>= 1) mx = fmaxf(mx, __shfl_xor(mx, o, 32));
    float sum = 0.0f;
#pragma unroll
    for (int i = 0; i < 16; ++i) {
        float e = expf(vals[i] - mx);   // exp(-inf - mx) -> 0
        vals[i] = e;
        sum += e;
    }
    for (int o = 16; o > 0; o >>= 1) sum += __shfl_xor(sum, o, 32);
    float inv = ind / sum;

    _Float16* avr = av_h + r * 512;
    float* outr = out_align + ((long long)qq * Bn + b) * nvalid;
#pragma unroll
    for (int i = 0; i < 16; ++i) {
        int j = lane + i * 32;
        float a = vals[i] * inv;
        avr[j] = (_Float16)a;
        if (j < nvalid) outr[j] = a;
    }
}

// ---------------- [B,Q,D] f32 -> [Q,B,D] f32 transpose ----------------
__global__ void k_transpose_bq(const float* __restrict__ x, float* __restrict__ y)
{
    long long t = (long long)blockIdx.x * blockDim.x + threadIdx.x;
    int d4 = (int)(t & 255);
    int qq = (int)((t >> 8) & 511);
    int b  = (int)(t >> 17);
    if (b >= Bn) return;
    float4 v = reinterpret_cast<const float4*>(x + ((long long)b * Qn + qq) * Dn)[d4];
    reinterpret_cast<float4*>(y + ((long long)qq * Bn + b) * Dn)[d4] = v;
}

// ============================ WMMA GEMM (async double-buffered) ============================
// C[M,N] (+)= A[M,K] * Bt^T ; A:[M,K] row-major, Bt:[N,K] row-major (pre-transposed).
// OMODE: 0 -> Cf = acc ; 1 -> Cf += acc ; 2 -> Ch = (f16)acc
// 128 threads = 4 waves, 64x64 C-tile, K-step 32. M,N multiples of 64, K of 32.
// Tiles streamed into LDS with global_load_async_to_lds_b128 (ASYNCcnt), double buffered.
#define TM 64
#define TN 64
#define TK 32
#define LSTR 48   // f16 elements per LDS row (32 data + 16 pad; 96B keeps b128 aligned)

template<int OMODE>
__global__ __launch_bounds__(128)
void k_gemm(const _Float16* __restrict__ A, int lda, long long sA,
            const _Float16* __restrict__ Bt, int ldb, long long sB,
            float* __restrict__ Cf, _Float16* __restrict__ Ch, int ldc, long long sC,
            int M, int N, int K)
{
    __shared__ _Float16 As[2][TM * LSTR];
    __shared__ _Float16 Bs[2][TN * LSTR];

    const int tid  = threadIdx.x;
    const int lane = tid & 31;
    const int wave = tid >> 5;
    const int waveM = wave >> 1;
    const int waveN = wave & 1;

    A  += (long long)blockIdx.z * sA;
    Bt += (long long)blockIdx.z * sB;
    const long long cOff = (long long)blockIdx.z * sC;

    const int tileM = blockIdx.y * TM;
    const int tileN = blockIdx.x * TN;

    const int row = tid >> 2;   // 0..31 (chunk rows; +32 for second chunk)
    const int c8  = tid & 3;    // 8-f16 column chunk within the 32-wide K tile

    v8f acc[2][2] = {};
    const int rsel = lane >> 4;   // K half-group select
    const int rlo  = lane & 15;

    // issue one 64x32 A tile + 64x32 B tile into buffer p via async copy (4 ops/thread)
    auto issue = [&](int p, int kt) {
#pragma unroll
        for (int it = 0; it < 2; ++it) {
            int r = row + it * 32;
            const _Float16* ga = A + (long long)(tileM + r) * lda + kt + c8 * 8;
            unsigned la = (unsigned)(unsigned long long)&As[p][r * LSTR + c8 * 8];
            asm volatile("global_load_async_to_lds_b128 %0, %1, off"
                         :: "v"(la), "v"((unsigned long long)ga) : "memory");
        }
#pragma unroll
        for (int it = 0; it < 2; ++it) {
            int r = row + it * 32;
            const _Float16* gb = Bt + (long long)(tileN + r) * ldb + kt + c8 * 8;
            unsigned lb = (unsigned)(unsigned long long)&Bs[p][r * LSTR + c8 * 8];
            asm volatile("global_load_async_to_lds_b128 %0, %1, off"
                         :: "v"(lb), "v"((unsigned long long)gb) : "memory");
        }
    };

    issue(0, 0);
    int p = 0;
    for (int kt = 0; kt < K; kt += TK, p ^= 1) {
        if (kt + TK < K) {
            issue(p ^ 1, kt + TK);                       // prefetch next tile into other buffer
            asm volatile("s_wait_asynccnt 4" ::: "memory");  // first 4 (current tile) complete
        } else {
            asm volatile("s_wait_asynccnt 0" ::: "memory");
        }
        __syncthreads();   // all waves' async data visible in LDS

        // gather fragments (ISA 16-bit A 16x32 layout; Bt rows are B^T rows)
        union FragU { uint4 u[2]; v16h v; };
        FragU a[2], b[2];
#pragma unroll
        for (int i = 0; i < 2; ++i) {
            const _Float16* ap = &As[p][(waveM * 32 + i * 16 + rlo) * LSTR + rsel * 8];
            a[i].u[0] = *reinterpret_cast<const uint4*>(ap);
            a[i].u[1] = *reinterpret_cast<const uint4*>(ap + 16);
            const _Float16* bp = &Bs[p][(waveN * 32 + i * 16 + rlo) * LSTR + rsel * 8];
            b[i].u[0] = *reinterpret_cast<const uint4*>(bp);
            b[i].u[1] = *reinterpret_cast<const uint4*>(bp + 16);
        }
#pragma unroll
        for (int i = 0; i < 2; ++i)
#pragma unroll
            for (int j = 0; j < 2; ++j)
                acc[i][j] = __builtin_amdgcn_wmma_f32_16x16x32_f16(
                    false, a[i].v, false, b[j].v, (short)0, acc[i][j], false, false);

        __syncthreads();   // done reading buffer p; it may be refilled next iteration
    }

    // ---- store C (8-VGPR f32 layout: lane%16 -> N, v + 8*(lane>>4) -> M) ----
    const int nBase = tileN + waveN * 32 + rlo;
    const int mBase = tileM + waveM * 32 + rsel * 8;
#pragma unroll
    for (int i = 0; i < 2; ++i)
#pragma unroll
        for (int j = 0; j < 2; ++j) {
            int n = nBase + j * 16;
#pragma unroll
            for (int v = 0; v < 8; ++v) {
                int m = mBase + i * 16 + v;
                long long idx = cOff + (long long)m * ldc + n;
                if (OMODE == 0)      Cf[idx] = acc[i][j][v];
                else if (OMODE == 1) Cf[idx] += acc[i][j][v];
                else                 Ch[idx] = (_Float16)acc[i][j][v];
            }
        }
}

template<int OM>
static void run_gemm(const _Float16* A, int lda, long long sA,
                     const _Float16* Bt, int ldb, long long sB,
                     float* Cf, _Float16* Ch, int ldc, long long sC,
                     int Mr, int Nr, int Kr, int batches, hipStream_t s)
{
    dim3 grid((unsigned)(Nr / 64), (unsigned)(Mr / 64), (unsigned)batches);
    k_gemm<OM><<<grid, dim3(128), 0, s>>>(A, lda, sA, Bt, ldb, sB, Cf, Ch, ldc, sC, Mr, Nr, Kr);
}

// ============================ driver ============================
extern "C" void kernel_launch(void* const* d_in, const int* in_sizes, int n_in,
                              void* d_out, int out_size, void* d_ws, size_t ws_size,
                              hipStream_t stream)
{
    (void)in_sizes; (void)n_in; (void)out_size; (void)ws_size;
    const float* q_f   = (const float*)d_in[0];
    const float* mem_f = (const float*)d_in[1];
    const int*   mlen  = (const int*)d_in[2];
    const float* wi0_f = (const float*)d_in[3];
    const float* wo0_f = (const float*)d_in[4];
    const float* wiN_f = (const float*)d_in[5];
    const float* woN_f = (const float*)d_in[6];
    float* out = (float*)d_out;

    // bump allocator over workspace (deterministic layout every call)
    char* base = (char*)d_ws;
    size_t off = 0;
    auto alloc = [&](size_t bytes) -> void* {
        void* p = base + off;
        off = (off + bytes + 255) & ~(size_t)255;
        return p;
    };

    _Float16* qh    = (_Float16*)alloc((size_t)Bn * Qn * Dn * 2);
    _Float16* mh    = (_Float16*)alloc((size_t)Bn * Mn * Dn * 2);
    _Float16* wi0h  = (_Float16*)alloc((size_t)Dn * Dn * 2);
    _Float16* wo0h  = (_Float16*)alloc((size_t)3 * Dn * Dn * 2);
    _Float16* wiNh  = (_Float16*)alloc((size_t)ORDERS * Dn * 2 * Dn * 2);
    _Float16* woNh  = (_Float16*)alloc((size_t)ORDERS * 2 * Dn * Dn * 2);
    _Float16* wi0T  = (_Float16*)alloc((size_t)Dn * Dn * 2);                 // [D,D]^T
    _Float16* wo0T  = (_Float16*)alloc((size_t)3 * Dn * Dn * 2);             // 3 blocks [D,D]^T
    _Float16* wiT   = (_Float16*)alloc((size_t)ORDERS * 2 * Dn * Dn * 2);    // [2D,D] per order
    _Float16* woT   = (_Float16*)alloc((size_t)ORDERS * Dn * 2 * Dn * 2);    // [D,2D] per order
    _Float16* mhT   = (_Float16*)alloc((size_t)Bn * Dn * Mn * 2);            // [B,D,M]
    _Float16* ng    = (_Float16*)alloc((size_t)Bn * 512 * 2 * Dn * 2);       // [B,512,2D]
    _Float16* ngT   = (_Float16*)alloc((size_t)Bn * 2 * Dn * 512 * 2);       // [B,2D,512]
    _Float16* qin   = (_Float16*)alloc((size_t)Bn * Qn * 2 * Dn * 2);
    float*    lg    = (float*)   alloc((size_t)Bn * Qn * 512 * 4);
    _Float16* gctx  = (_Float16*)alloc((size_t)Bn * Qn * Dn * 2);
    float*    ngsum = (float*)   alloc((size_t)Bn * Qn * Dn * 4);
    _Float16* ngctx = (_Float16*)alloc((size_t)Bn * Qn * Dn * 2);
    void*     R0    = alloc((size_t)Bn * Qn * 2 * Dn * 2);  // shared lifetimes: qin0 -> c -> attn
    _Float16* qin0  = (_Float16*)R0;
    _Float16* ch    = (_Float16*)R0;
    float*    attn  = (float*)R0;
    _Float16* av    = qin;   // alias: av written only after qin consumed each stage

    // output slice offsets (return order: attn_h, g_av, a1, a2, a3)
    float* out_attn = out;
    float* out_gav  = out_attn + (size_t)Qn * Bn * Dn;
    float* out_a1   = out_gav  + (size_t)Qn * Bn * 512;
    float* out_a2   = out_a1   + (size_t)Qn * Bn * 512;   // order1: nvalid=512
    float* out_a3   = out_a2   + (size_t)Qn * Bn * 511;   // order2: nvalid=511
    float* outs[3]  = { out_a1, out_a2, out_a3 };

    // ---- (1) convert operands to f16, then pre-transpose to [N,K] layouts ----
    auto cvt = [&](const float* src, _Float16* dst, long long n) {
        long long n4 = n >> 2;
        k_cvt_f16<<<dim3((unsigned)((n4 + 255) / 256)), dim3(256), 0, stream>>>(src, dst, n4);
    };
    cvt(q_f,   qh,   (long long)Bn * Qn * Dn);
    cvt(mem_f, mh,   (long long)Bn * Mn * Dn);
    cvt(wi0_f, wi0h, (long long)Dn * Dn);
    cvt(wo0_f, wo0h, (long long)3 * Dn * Dn);
    cvt(wiN_f, wiNh, (long long)ORDERS * Dn * 2 * Dn);
    cvt(woN_f, woNh, (long long)ORDERS * 2 * Dn * Dn);

    run_tr(wi0h, wi0T, Dn, Dn, 1, stream);            // W_in0^T
    run_tr(wo0h, wo0T, Dn, Dn, 3, stream);            // each D-row block of W_out0, transposed
    run_tr(wiNh, wiT, Dn, 2 * Dn, ORDERS, stream);    // W_in[o]^T  : [2D, D]
    run_tr(woNh, woT, 2 * Dn, Dn, ORDERS, stream);    // W_out[o]^T : [D, 2D]
    run_tr(mh,   mhT, Mn, Dn, Bn, stream);            // mem^T per batch : [D, M]

    const dim3 smg(Bn * Qn / 8), smb(256);
    const dim3 ewg(Bn * 512 * (Dn / 4) / 256), ewb(256);

    // ---- (2) qin0 = query @ W_in0   [16384 x 1024, K=1024] ----
    run_gemm<2>(qh, Dn, 0, wi0T, Dn, 0, nullptr, qin0, Dn, 0,
                Bn * Qn, Dn, Dn, 1, stream);
    // ---- (3) g logits = qin0 @ mem^T (per batch; mem rows already [N=M, K=D]) ----
    run_gemm<0>(qin0, Dn, (long long)Qn * Dn, mh, Dn, (long long)Mn * Dn,
                lg, nullptr, Mn, (long long)Qn * Mn, Qn, Mn, Dn, Bn, stream);
    // ---- (4) masked softmax (order 0) -> g_av output + f16 av ----
    k_softmax<<<smg, smb, 0, stream>>>(lg, mlen, out_gav, av, 0, Mn);
    // ---- (5) g_ctx = av @ mem (per batch; B^T = mhT [D,M]) ----
    run_gemm<2>(av, Mn, (long long)Qn * Mn, mhT, Mn, (long long)Dn * Mn,
                nullptr, gctx, Dn, (long long)Qn * Dn, Qn, Dn, Mn, Bn, stream);

    // ---- (6) n-gram orders ----
    for (int o = 1; o <= ORDERS; ++o) {
        int nvalid = Mn - o + 1;
        // (a) build padded ngram embedding [B,512,2D] and its transpose [B,2D,512]
        k_ngram_build<<<ewg, ewb, 0, stream>>>(mem_f, ng, o, nvalid);
        run_tr(ng, ngT, 512, 2 * Dn, Bn, stream);
        // (b) qin = query @ W_in[o-1]   [16384 x 2048, K=1024]
        run_gemm<2>(qh, Dn, 0, wiT + (long long)(o - 1) * 2 * Dn * Dn, Dn, 0,
                    nullptr, qin, 2 * Dn, 0, Bn * Qn, 2 * Dn, Dn, 1, stream);
        // (c) logits = qin @ ng^T (per batch; ng rows are [N=512, K=2D])
        run_gemm<0>(qin, 2 * Dn, (long long)Qn * 2 * Dn, ng, 2 * Dn, (long long)512 * 2 * Dn,
                    lg, nullptr, 512, (long long)Qn * 512, Qn, 512, 2 * Dn, Bn, stream);
        // (d) masked softmax + indicator -> align output + f16 av (zero padded)
        k_softmax<<<smg, smb, 0, stream>>>(lg, mlen, outs[o - 1], av, o, nvalid);
        // (e) c = av @ ng (per batch; B^T = ngT [2D, 512])
        run_gemm<2>(av, 512, (long long)Qn * 512, ngT, 512, (long long)2 * Dn * 512,
                    nullptr, ch, 2 * Dn, (long long)Qn * 2 * Dn, Qn, 2 * Dn, 512, Bn, stream);
        // (f) ngram_sum (+)= c @ W_out[o-1]   [16384 x 1024, K=2048]
        if (o == 1)
            run_gemm<0>(ch, 2 * Dn, 0, woT, 2 * Dn, 0, ngsum, nullptr, Dn, 0,
                        Bn * Qn, Dn, 2 * Dn, 1, stream);
        else
            run_gemm<1>(ch, 2 * Dn, 0, woT + (long long)(o - 1) * Dn * 2 * Dn, 2 * Dn, 0,
                        ngsum, nullptr, Dn, 0, Bn * Qn, Dn, 2 * Dn, 1, stream);
    }

    // ---- (7) ngram_ctx = tanh(ngram_sum) -> f16 ----
    {
        long long n4 = (long long)Bn * Qn * Dn / 4;
        k_tanh_f16<<<dim3((unsigned)((n4 + 255) / 256)), dim3(256), 0, stream>>>(ngsum, ngctx, n4);
    }

    // ---- (8) attn = query@W0[0:D] + g_ctx@W0[D:2D] + ngram_ctx@W0[2D:3D] ----
    run_gemm<0>(qh,    Dn, 0, wo0T,                          Dn, 0, attn, nullptr, Dn, 0, Bn * Qn, Dn, Dn, 1, stream);
    run_gemm<1>(gctx,  Dn, 0, wo0T + (long long)Dn * Dn,     Dn, 0, attn, nullptr, Dn, 0, Bn * Qn, Dn, Dn, 1, stream);
    run_gemm<1>(ngctx, Dn, 0, wo0T + (long long)2 * Dn * Dn, Dn, 0, attn, nullptr, Dn, 0, Bn * Qn, Dn, Dn, 1, stream);

    // ---- (9) [B,Q,D] -> [Q,B,D] transposed output ----
    k_transpose_bq<<<ewg, ewb, 0, stream>>>(attn, out_attn);
}
